// PatchTSTModel_34780645163679
// MI455X (gfx1250) — matrix operand order repaired and checked by reading, not compiled, not run on hardware
//
#include <hip/hip_runtime.h>
#include <math.h>

// ---------------- model constants ----------------
#define BS        32
#define LOOKBACK  512
#define FEATURES  128
#define ENC_IN    64
#define D_MODEL   512
#define N_HEADS   8
#define HEAD_DIM  64
#define D_FF      2048
#define E_LAYERS  3
#define PATCH_LEN 16
#define STRIDE    8
#define NPATCH    64
#define BFLAT     (BS * ENC_IN)          // 2048 sequences
#define MTOK      (BFLAT * NPATCH)       // 131072 tokens
#define EPS       1e-5f

// Use CDNA5 async global->LDS DMA (GLOBAL_LOAD_ASYNC_TO_LDS_B128, ASYNCcnt)
#define USE_ASYNC_COPY 1

typedef __attribute__((ext_vector_type(16))) _Float16 v16h;
typedef __attribute__((ext_vector_type(8)))  float    v8f;

union F16Frag { v16h v; float4 q[2]; };

static __device__ inline v8f vzero8() {
    v8f z = {0.f, 0.f, 0.f, 0.f, 0.f, 0.f, 0.f, 0.f};
    return z;
}

#if USE_ASYNC_COPY
// Async copy of 16 bytes global -> LDS. dst must be an LDS byte address
// (low 32 bits of a generic pointer to __shared__), src a global pointer.
static __device__ inline void async_copy_b128(const _Float16* gsrc, void* lds_dst) {
    unsigned lds_addr = (unsigned)(unsigned long long)lds_dst;  // low 32 bits = LDS offset
    asm volatile("global_load_async_to_lds_b128 %0, %1, off"
                 :: "v"(lds_addr), "v"(gsrc)
                 : "memory");
}
// Wait until at most `n` async ops remain outstanding for this wave.
// Async loads complete in order, so with 3 ops/tile, n=3 waits for the
// previous tile while the next tile's DMA stays in flight.
static __device__ inline void async_wait_le3() {
    asm volatile("s_wait_asynccnt 3" ::: "memory");
}
static __device__ inline void async_wait_all() {
    asm volatile("s_wait_asynccnt 0" ::: "memory");
}
#endif

// =====================================================================
// Weight convert+transpose: dst[n*K + k] = (f16) src[k*N + n]
// =====================================================================
__global__ void k_cvt_t(_Float16* __restrict__ dst, const float* __restrict__ src,
                        int K, int N) {
    long total = (long)K * N;
    for (long idx = (long)blockIdx.x * blockDim.x + threadIdx.x; idx < total;
         idx += (long)gridDim.x * blockDim.x) {
        long n = idx / K, k = idx - n * K;
        dst[idx] = (_Float16)src[k * (long)N + n];
    }
}

__global__ void k_copy(float* __restrict__ dst, const float* __restrict__ src, int n) {
    for (int i = blockIdx.x * blockDim.x + threadIdx.x; i < n; i += gridDim.x * blockDim.x)
        dst[i] = src[i];
}

__global__ void k_zero(float* __restrict__ p, int n) {
    for (int i = blockIdx.x * blockDim.x + threadIdx.x; i < n; i += gridDim.x * blockDim.x)
        p[i] = 0.f;
}

// =====================================================================
// Entry projection: xe[b,l,:] = x[b,l,:] @ entry_w + entry_b
// grid = BS*LOOKBACK blocks, 64 threads
// =====================================================================
__global__ __launch_bounds__(64)
void k_entry(const float* __restrict__ x, const float* __restrict__ ew,
             const float* __restrict__ eb, float* __restrict__ xe) {
    __shared__ float row[FEATURES];
    long bl = blockIdx.x;
    int t = threadIdx.x;
    row[t]      = x[bl * FEATURES + t];
    row[t + 64] = x[bl * FEATURES + t + 64];
    __syncthreads();
    float acc = eb[t];
    #pragma unroll 8
    for (int k = 0; k < FEATURES; ++k) acc += row[k] * ew[k * ENC_IN + t];
    xe[bl * ENC_IN + t] = acc;
}

// =====================================================================
// Per-(b,channel) mean / stdev over L. grid = BFLAT, 256 threads
// =====================================================================
__global__ __launch_bounds__(256)
void k_stats(const float* __restrict__ xe, float* __restrict__ means,
             float* __restrict__ stdev) {
    __shared__ float s1[256], s2[256];
    int bc = blockIdx.x, b = bc >> 6, c = bc & 63;
    int t = threadIdx.x;
    const float* p = xe + (long)b * LOOKBACK * ENC_IN + c;
    float v0 = p[(long)t * ENC_IN];
    float v1 = p[(long)(t + 256) * ENC_IN];
    s1[t] = v0 + v1;
    s2[t] = v0 * v0 + v1 * v1;
    __syncthreads();
    for (int s = 128; s > 0; s >>= 1) {
        if (t < s) { s1[t] += s1[t + s]; s2[t] += s2[t + s]; }
        __syncthreads();
    }
    if (t == 0) {
        float mean = s1[0] / (float)LOOKBACK;
        float var  = s2[0] / (float)LOOKBACK - mean * mean;
        means[bc] = mean;
        stdev[bc] = sqrtf(var + EPS);
    }
}

// =====================================================================
// Patch embed + positional embedding -> X (f32) and Xh (f16)
// grid = MTOK, 256 threads. Replication-pad via index clamp.
// =====================================================================
__global__ __launch_bounds__(256)
void k_patch(const float* __restrict__ xe, const float* __restrict__ means,
             const float* __restrict__ stdev, const float* __restrict__ val_w,
             float* __restrict__ X, _Float16* __restrict__ Xh) {
    __shared__ float pv[PATCH_LEN];
    long m = blockIdx.x;
    int bc = (int)(m >> 6), p = (int)(m & 63);
    int b = bc >> 6, c = bc & 63;
    int t = threadIdx.x;
    if (t < PATCH_LEN) {
        int l = p * STRIDE + t;
        if (l > LOOKBACK - 1) l = LOOKBACK - 1;
        pv[t] = (xe[((long)b * LOOKBACK + l) * ENC_IN + c] - means[bc]) / stdev[bc];
    }
    __syncthreads();
    for (int d = t; d < D_MODEL; d += 256) {
        float acc = 0.f;
        #pragma unroll
        for (int j = 0; j < PATCH_LEN; ++j) acc += pv[j] * val_w[j * D_MODEL + d];
        // pos emb: pe[p][2i]=sin(p*div_i), pe[p][2i+1]=cos, div_i=exp(-2i*ln(1e4)/512)
        float ang = (float)p * expf((float)(d & ~1) * (-9.210340371976184f / (float)D_MODEL));
        acc += (d & 1) ? cosf(ang) : sinf(ang);
        X[m * D_MODEL + d]  = acc;
        Xh[m * D_MODEL + d] = (_Float16)acc;
    }
}

// =====================================================================
// WMMA GEMM: C[M,N] = A[M,K](f16) @ Wt[N,K](f16)^T (+bias)(+gelu)(+res)
// block = 256 thr = 8 waves, tile 128x64, KC=32, double-buffered LDS.
// Tile staging: GLOBAL_LOAD_ASYNC_TO_LDS_B128 (3 ops/tile/wave, in-order
// completion) so s_wait_asynccnt 3 overlaps tile k+1 DMA with tile k WMMA.
// Fragment layouts per CDNA5 ISA 7.12.2 (wave32).
// =====================================================================
__global__ __launch_bounds__(256)
void k_gemm(const _Float16* __restrict__ A, const _Float16* __restrict__ Wt,
            const float* __restrict__ bias, const float* __restrict__ res,
            float* __restrict__ Cf, _Float16* __restrict__ Ch,
            int N, int K, int gelu) {
    __shared__ _Float16 Ah[2][128 * 32];
    __shared__ _Float16 Bh[2][64 * 32];
    const int t = threadIdx.x;
    const int lane = t & 31, w = t >> 5;
    const int wm = w >> 1, wn = w & 1;       // 4x2 wave grid
    const int kh = lane >> 4, lm = lane & 15;
    const long m0 = (long)blockIdx.x * 128;
    const long n0 = (long)blockIdx.y * 64;

    v8f acc[2][2];
    acc[0][0] = vzero8(); acc[0][1] = vzero8();
    acc[1][0] = vzero8(); acc[1][1] = vzero8();

    auto stage_tile = [&](int k0, int buf) {
#if USE_ASYNC_COPY
        #pragma unroll
        for (int c = 0; c < 2; ++c) {                 // A tile 128x32: 2x16B/thr
            int idx = t + c * 256;
            int row = idx >> 2, col = (idx & 3) * 8;
            async_copy_b128(&A[(m0 + row) * (long)K + k0 + col],
                            &Ah[buf][row * 32 + col]);
        }
        {                                              // B tile 64x32: 1x16B/thr
            int row = t >> 2, col = (t & 3) * 8;
            async_copy_b128(&Wt[(n0 + row) * (long)K + k0 + col],
                            &Bh[buf][row * 32 + col]);
        }
#else
        #pragma unroll
        for (int c = 0; c < 2; ++c) {
            int idx = t + c * 256;
            int row = idx >> 2, col = (idx & 3) * 8;
            *reinterpret_cast<float4*>(&Ah[buf][row * 32 + col]) =
                *reinterpret_cast<const float4*>(&A[(m0 + row) * (long)K + k0 + col]);
        }
        {
            int row = t >> 2, col = (t & 3) * 8;
            *reinterpret_cast<float4*>(&Bh[buf][row * 32 + col]) =
                *reinterpret_cast<const float4*>(&Wt[(n0 + row) * (long)K + k0 + col]);
        }
#endif
    };

    const int ntiles = K >> 5;
    stage_tile(0, 0);                                  // prologue

    for (int i = 0; i < ntiles; ++i) {
        const int buf = i & 1;
        if (i + 1 < ntiles) {
            stage_tile((i + 1) << 5, buf ^ 1);         // DMA next tile, other buffer
            if (i + 2 < ntiles) {                      // prefetch 2 tiles ahead
                __builtin_prefetch(&A[(m0 + (t >> 1)) * (long)K + ((i + 2) << 5)], 0, 1);
                __builtin_prefetch(&Wt[(n0 + (t & 63)) * (long)K + ((i + 2) << 5)], 0, 1);
            }
#if USE_ASYNC_COPY
            async_wait_le3();                          // tile i done, i+1 in flight
#endif
        } else {
#if USE_ASYNC_COPY
            async_wait_all();
#endif
        }
        __syncthreads();                               // tile i visible to all waves

        F16Frag a[2], b[2];
        #pragma unroll
        for (int mt = 0; mt < 2; ++mt) {
            int arow = wm * 32 + mt * 16 + lm;
            a[mt].q[0] = *reinterpret_cast<const float4*>(&Ah[buf][arow * 32 + kh * 8]);
            a[mt].q[1] = *reinterpret_cast<const float4*>(&Ah[buf][arow * 32 + 16 + kh * 8]);
        }
        #pragma unroll
        for (int nt = 0; nt < 2; ++nt) {
            int brow = wn * 32 + nt * 16 + lm;
            b[nt].q[0] = *reinterpret_cast<const float4*>(&Bh[buf][brow * 32 + kh * 8]);
            b[nt].q[1] = *reinterpret_cast<const float4*>(&Bh[buf][brow * 32 + 16 + kh * 8]);
        }
        #pragma unroll
        for (int mt = 0; mt < 2; ++mt)
            #pragma unroll
            for (int nt = 0; nt < 2; ++nt)
                acc[mt][nt] = __builtin_amdgcn_wmma_f32_16x16x32_f16(
                    false, a[mt].v, false, b[nt].v, (short)0, acc[mt][nt], false, false);
        __syncthreads();                               // guard buf reuse at i+1's stage
    }

    // epilogue: C/D layout -> lane(l): N=l%16, M=(l/16)*8+r
    #pragma unroll
    for (int mt = 0; mt < 2; ++mt)
        #pragma unroll
        for (int nt = 0; nt < 2; ++nt) {
            long gn = n0 + wn * 32 + nt * 16 + lm;
            float bv = bias ? bias[gn] : 0.f;
            #pragma unroll
            for (int r = 0; r < 8; ++r) {
                long gm = m0 + wm * 32 + mt * 16 + kh * 8 + r;
                float v = acc[mt][nt][r] + bv;
                if (gelu) v = 0.5f * v * (1.0f + erff(v * 0.70710678118654752f));
                if (res)  v += res[gm * N + gn];
                if (Cf) Cf[gm * N + gn] = v;
                if (Ch) Ch[gm * N + gn] = (_Float16)v;
            }
        }
}

// =====================================================================
// LayerNorm (in place) -> X (f32) and Xh (f16). grid = MTOK, 256 thr.
// =====================================================================
__global__ __launch_bounds__(256)
void k_ln(float* __restrict__ X, _Float16* __restrict__ Xh,
          const float* __restrict__ g, const float* __restrict__ b) {
    __shared__ float s1[256], s2[256];
    long m = blockIdx.x;
    int t = threadIdx.x;
    float v0 = X[m * D_MODEL + t];
    float v1 = X[m * D_MODEL + t + 256];
    s1[t] = v0 + v1;
    s2[t] = v0 * v0 + v1 * v1;
    __syncthreads();
    for (int s = 128; s > 0; s >>= 1) {
        if (t < s) { s1[t] += s1[t + s]; s2[t] += s2[t + s]; }
        __syncthreads();
    }
    float mean = s1[0] / (float)D_MODEL;
    float var  = s2[0] / (float)D_MODEL - mean * mean;
    float rstd = rsqrtf(var + EPS);
    float o0 = (v0 - mean) * rstd * g[t] + b[t];
    float o1 = (v1 - mean) * rstd * g[t + 256] + b[t + 256];
    X[m * D_MODEL + t]        = o0;
    X[m * D_MODEL + t + 256]  = o1;
    Xh[m * D_MODEL + t]       = (_Float16)o0;
    Xh[m * D_MODEL + t + 256] = (_Float16)o1;
}

// =====================================================================
// Attention per (batch, head): S = Q K^T / 8, softmax, O = P V (all WMMA)
// grid = (BFLAT, N_HEADS), block = 256 thr.
// =====================================================================
__global__ __launch_bounds__(256)
void k_attn(const _Float16* __restrict__ QKV, _Float16* __restrict__ Oh) {
    __shared__ _Float16 Qs[64 * 64];
    __shared__ _Float16 Ks[64 * 64];
    __shared__ _Float16 Vt[64 * 64];      // transposed: Vt[d][seq]
    __shared__ float    Ss[64 * 64];
    __shared__ _Float16 Ps[64 * 64];
    const int Bi = blockIdx.x, h = blockIdx.y;
    const long m0 = (long)Bi * NPATCH;
    const int t = threadIdx.x, lane = t & 31, w = t >> 5;
    const int kh = lane >> 4, lm = lane & 15;
    const int tr = w >> 1;                 // seq tile row 0..3
    const long QSTRIDE = 3 * D_MODEL;      // 1536

    for (int c = t; c < 512; c += 256) {   // Q,K tiles (8 f16 per chunk)
        int i = c >> 3, off = (c & 7) * 8;
        long rb = (m0 + i) * QSTRIDE + (long)h * HEAD_DIM + off;
#if USE_ASYNC_COPY
        async_copy_b128(&QKV[rb],           &Qs[i * 64 + off]);
        async_copy_b128(&QKV[rb + D_MODEL], &Ks[i * 64 + off]);
#else
        *reinterpret_cast<float4*>(&Qs[i * 64 + off]) =
            *reinterpret_cast<const float4*>(&QKV[rb]);
        *reinterpret_cast<float4*>(&Ks[i * 64 + off]) =
            *reinterpret_cast<const float4*>(&QKV[rb + D_MODEL]);
#endif
    }
    for (int idx = t; idx < 4096; idx += 256) {   // V transposed (scalar)
        int i = idx >> 6, d = idx & 63;
        Vt[d * 64 + i] = QKV[(m0 + i) * QSTRIDE + 2 * D_MODEL + (long)h * HEAD_DIM + d];
    }
#if USE_ASYNC_COPY
    async_wait_all();
#endif
    __syncthreads();

    // ---- stage 1: S = Q @ K^T * 0.125 ----
    {
        v8f acc[2] = {vzero8(), vzero8()};
        #pragma unroll
        for (int kc = 0; kc < 2; ++kc) {
            F16Frag a;
            int arow = tr * 16 + lm;
            a.q[0] = *reinterpret_cast<const float4*>(&Qs[arow * 64 + kc * 32 + kh * 8]);
            a.q[1] = *reinterpret_cast<const float4*>(&Qs[arow * 64 + kc * 32 + 16 + kh * 8]);
            #pragma unroll
            for (int tcb = 0; tcb < 2; ++tcb) {
                int tc = (w & 1) * 2 + tcb;
                F16Frag b;
                int brow = tc * 16 + lm;
                b.q[0] = *reinterpret_cast<const float4*>(&Ks[brow * 64 + kc * 32 + kh * 8]);
                b.q[1] = *reinterpret_cast<const float4*>(&Ks[brow * 64 + kc * 32 + 16 + kh * 8]);
                acc[tcb] = __builtin_amdgcn_wmma_f32_16x16x32_f16(
                    false, a.v, false, b.v, (short)0, acc[tcb], false, false);
            }
        }
        #pragma unroll
        for (int tcb = 0; tcb < 2; ++tcb) {
            int tc = (w & 1) * 2 + tcb;
            #pragma unroll
            for (int r = 0; r < 8; ++r)
                Ss[(tr * 16 + kh * 8 + r) * 64 + tc * 16 + lm] = acc[tcb][r] * 0.125f;
        }
    }
    __syncthreads();

    // ---- softmax rows (thread t < 64 owns row t) ----
    if (t < 64) {
        float mx = -1e30f;
        for (int j = 0; j < 64; ++j) mx = fmaxf(mx, Ss[t * 64 + j]);
        float sum = 0.f;
        for (int j = 0; j < 64; ++j) {
            float e = expf(Ss[t * 64 + j] - mx);
            Ss[t * 64 + j] = e;
            sum += e;
        }
        float inv = 1.f / sum;
        for (int j = 0; j < 64; ++j) Ps[t * 64 + j] = (_Float16)(Ss[t * 64 + j] * inv);
    }
    __syncthreads();

    // ---- stage 2: O = P @ V  (Bt = Vt) ----
    {
        v8f acc[2] = {vzero8(), vzero8()};
        #pragma unroll
        for (int kc = 0; kc < 2; ++kc) {
            F16Frag a;
            int arow = tr * 16 + lm;
            a.q[0] = *reinterpret_cast<const float4*>(&Ps[arow * 64 + kc * 32 + kh * 8]);
            a.q[1] = *reinterpret_cast<const float4*>(&Ps[arow * 64 + kc * 32 + 16 + kh * 8]);
            #pragma unroll
            for (int tcb = 0; tcb < 2; ++tcb) {
                int tc = (w & 1) * 2 + tcb;
                F16Frag b;
                int brow = tc * 16 + lm;
                b.q[0] = *reinterpret_cast<const float4*>(&Vt[brow * 64 + kc * 32 + kh * 8]);
                b.q[1] = *reinterpret_cast<const float4*>(&Vt[brow * 64 + kc * 32 + 16 + kh * 8]);
                acc[tcb] = __builtin_amdgcn_wmma_f32_16x16x32_f16(
                    false, a.v, false, b.v, (short)0, acc[tcb], false, false);
            }
        }
        #pragma unroll
        for (int tcb = 0; tcb < 2; ++tcb) {
            int tc = (w & 1) * 2 + tcb;
            #pragma unroll
            for (int r = 0; r < 8; ++r) {
                int seq = tr * 16 + kh * 8 + r, dc = tc * 16 + lm;
                Oh[(m0 + seq) * D_MODEL + (long)h * HEAD_DIM + dc] = (_Float16)acc[tcb][r];
            }
        }
    }
}

// =====================================================================
// BatchNorm stats over all tokens per channel (thread owns 2 columns)
// grid = MTOK/128, 256 thr
// =====================================================================
__global__ __launch_bounds__(256)
void k_bnstats(const float* __restrict__ X, float* __restrict__ sums,
               float* __restrict__ sqs) {
    long r0 = (long)blockIdx.x * 128;
    int t = threadIdx.x;
    float s0 = 0, s1 = 0, q0 = 0, q1 = 0;
    for (int r = 0; r < 128; ++r) {
        float a = X[(r0 + r) * D_MODEL + t];
        float b = X[(r0 + r) * D_MODEL + t + 256];
        s0 += a; q0 += a * a; s1 += b; q1 += b * b;
    }
    atomicAdd(&sums[t], s0);       atomicAdd(&sums[t + 256], s1);
    atomicAdd(&sqs[t], q0);        atomicAdd(&sqs[t + 256], q1);
}

__global__ void k_bnfin(const float* __restrict__ sums, const float* __restrict__ sqs,
                        const float* __restrict__ g, const float* __restrict__ b,
                        float* __restrict__ scale, float* __restrict__ shift) {
    int d = blockIdx.x * blockDim.x + threadIdx.x;
    if (d >= D_MODEL) return;
    float mean = sums[d] / (float)MTOK;
    float var  = sqs[d] / (float)MTOK - mean * mean;
    float sc = rsqrtf(var + EPS) * g[d];
    scale[d] = sc;
    shift[d] = b[d] - mean * sc;
}

// =====================================================================
// Flatten head + denormalise: val[bc] = (sum_{p,d} BN(x)[bc,p,d]*h1w[d*64+p] + h1b)*std+mean
// grid = BFLAT, 256 thr
// =====================================================================
__global__ __launch_bounds__(256)
void k_head(const float* __restrict__ X, const float* __restrict__ scale,
            const float* __restrict__ shift, const float* __restrict__ h1w,
            const float* __restrict__ h1b, const float* __restrict__ means,
            const float* __restrict__ stdev, float* __restrict__ val) {
    __shared__ float red[256];
    int bc = blockIdx.x, t = threadIdx.x;
    long base = (long)bc * NPATCH * D_MODEL;
    float acc = 0.f;
    for (int idx = t; idx < NPATCH * D_MODEL; idx += 256) {
        int p = idx >> 9, d = idx & 511;
        float xv = X[base + (long)p * D_MODEL + d] * scale[d] + shift[d];
        acc += xv * h1w[d * NPATCH + p];
    }
    red[t] = acc;
    __syncthreads();
    for (int s = 128; s > 0; s >>= 1) {
        if (t < s) red[t] += red[t + s];
        __syncthreads();
    }
    if (t == 0) val[bc] = (red[0] + h1b[0]) * stdev[bc] + means[bc];
}

__global__ __launch_bounds__(64)
void k_final(const float* __restrict__ val, const float* __restrict__ h2w,
             const float* __restrict__ h2b, float* __restrict__ out) {
    __shared__ float red[64];
    int b = blockIdx.x, t = threadIdx.x;
    red[t] = val[b * ENC_IN + t] * h2w[t];
    __syncthreads();
    for (int s = 32; s > 0; s >>= 1) {
        if (t < s) red[t] += red[t + s];
        __syncthreads();
    }
    if (t == 0) out[b] = red[0] + h2b[0];
}

// =====================================================================
// launch
// =====================================================================
extern "C" void kernel_launch(void* const* d_in, const int* in_sizes, int n_in,
                              void* d_out, int out_size, void* d_ws, size_t ws_size,
                              hipStream_t stream) {
    const float* x       = (const float*)d_in[0];
    const float* entry_w = (const float*)d_in[1];
    const float* entry_b = (const float*)d_in[2];
    const float* val_w   = (const float*)d_in[3];
    const float* wq = (const float*)d_in[4];  const float* bq = (const float*)d_in[5];
    const float* wk = (const float*)d_in[6];  const float* bk = (const float*)d_in[7];
    const float* wv = (const float*)d_in[8];  const float* bv = (const float*)d_in[9];
    const float* wo = (const float*)d_in[10]; const float* bo = (const float*)d_in[11];
    const float* c1w = (const float*)d_in[12]; const float* c1b = (const float*)d_in[13];
    const float* c2w = (const float*)d_in[14]; const float* c2b = (const float*)d_in[15];
    const float* ln1g = (const float*)d_in[16]; const float* ln1b = (const float*)d_in[17];
    const float* ln2g = (const float*)d_in[18]; const float* ln2b = (const float*)d_in[19];
    const float* bng = (const float*)d_in[20]; const float* bnb = (const float*)d_in[21];
    const float* h1w = (const float*)d_in[22]; const float* h1b = (const float*)d_in[23];
    const float* h2w = (const float*)d_in[24]; const float* h2b = (const float*)d_in[25];
    float* out = (float*)d_out;

    // ---------------- workspace carve-up ----------------
    size_t off = 0;
    auto carve = [&](size_t bytes) -> void* {
        void* p = (char*)d_ws + off;
        off += (bytes + 255) & ~(size_t)255;
        return p;
    };
    _Float16* Wt_qkv = (_Float16*)carve((size_t)E_LAYERS * 3 * D_MODEL * D_MODEL * 2);
    _Float16* Wt_o   = (_Float16*)carve((size_t)E_LAYERS * D_MODEL * D_MODEL * 2);
    _Float16* Wt_c1  = (_Float16*)carve((size_t)E_LAYERS * D_FF * D_MODEL * 2);
    _Float16* Wt_c2  = (_Float16*)carve((size_t)E_LAYERS * D_MODEL * D_FF * 2);
    float* bqkv  = (float*)carve((size_t)E_LAYERS * 3 * D_MODEL * 4);
    float* means = (float*)carve(BFLAT * 4);
    float* stdev = (float*)carve(BFLAT * 4);
    float* xe    = (float*)carve((size_t)BS * LOOKBACK * ENC_IN * 4);
    float* X     = (float*)carve((size_t)MTOK * D_MODEL * 4);
    _Float16* Xh = (_Float16*)carve((size_t)MTOK * D_MODEL * 2);
    _Float16* U  = (_Float16*)carve((size_t)MTOK * D_FF * 2);  // time-shared region
    _Float16* QKVh = U;                                        // [MTOK,1536]
    _Float16* Oh   = U + (size_t)MTOK * 3 * D_MODEL;           // [MTOK,512]
    _Float16* Yh   = U;                                        // [MTOK,2048]
    float* bns   = (float*)carve(D_MODEL * 4);
    float* bnq   = (float*)carve(D_MODEL * 4);
    float* bnsc  = (float*)carve(D_MODEL * 4);
    float* bnsh  = (float*)carve(D_MODEL * 4);
    float* valb  = (float*)carve(BFLAT * 4);
    (void)ws_size; (void)in_sizes; (void)n_in; (void)out_size;

    // ---------------- weight prep (f16, [N][K]) ----------------
    for (int i = 0; i < E_LAYERS; ++i) {
        size_t wsz = (size_t)D_MODEL * D_MODEL;
        k_cvt_t<<<1024, 256, 0, stream>>>(Wt_qkv + i * 3 * wsz + 0 * wsz, wq + i * wsz, D_MODEL, D_MODEL);
        k_cvt_t<<<1024, 256, 0, stream>>>(Wt_qkv + i * 3 * wsz + 1 * wsz, wk + i * wsz, D_MODEL, D_MODEL);
        k_cvt_t<<<1024, 256, 0, stream>>>(Wt_qkv + i * 3 * wsz + 2 * wsz, wv + i * wsz, D_MODEL, D_MODEL);
        k_cvt_t<<<1024, 256, 0, stream>>>(Wt_o + i * wsz, wo + i * wsz, D_MODEL, D_MODEL);
        k_cvt_t<<<4096, 256, 0, stream>>>(Wt_c1 + (size_t)i * D_FF * D_MODEL,
                                          c1w + (size_t)i * D_MODEL * D_FF, D_MODEL, D_FF);
        k_cvt_t<<<4096, 256, 0, stream>>>(Wt_c2 + (size_t)i * D_MODEL * D_FF,
                                          c2w + (size_t)i * D_FF * D_MODEL, D_FF, D_MODEL);
        k_copy<<<2, 256, 0, stream>>>(bqkv + i * 3 * D_MODEL + 0 * D_MODEL, bq + i * D_MODEL, D_MODEL);
        k_copy<<<2, 256, 0, stream>>>(bqkv + i * 3 * D_MODEL + 1 * D_MODEL, bk + i * D_MODEL, D_MODEL);
        k_copy<<<2, 256, 0, stream>>>(bqkv + i * 3 * D_MODEL + 2 * D_MODEL, bv + i * D_MODEL, D_MODEL);
    }

    // ---------------- front end ----------------
    k_entry<<<BS * LOOKBACK, 64, 0, stream>>>(x, entry_w, entry_b, xe);
    k_stats<<<BFLAT, 256, 0, stream>>>(xe, means, stdev);
    k_patch<<<MTOK, 256, 0, stream>>>(xe, means, stdev, val_w, X, Xh);

    // ---------------- encoder layers ----------------
    for (int i = 0; i < E_LAYERS; ++i) {
        size_t wsz = (size_t)D_MODEL * D_MODEL;
        // QKV: [MTOK,512] x [512,1536]
        k_gemm<<<dim3(MTOK / 128, (3 * D_MODEL) / 64), 256, 0, stream>>>(
            Xh, Wt_qkv + i * 3 * wsz, bqkv + i * 3 * D_MODEL,
            nullptr, nullptr, QKVh, 3 * D_MODEL, D_MODEL, 0);
        // attention
        k_attn<<<dim3(BFLAT, N_HEADS), 256, 0, stream>>>(QKVh, Oh);
        // O-proj + residual (in place into X)
        k_gemm<<<dim3(MTOK / 128, D_MODEL / 64), 256, 0, stream>>>(
            Oh, Wt_o + i * wsz, bo + i * D_MODEL, X, X, nullptr, D_MODEL, D_MODEL, 0);
        k_ln<<<MTOK, 256, 0, stream>>>(X, Xh, ln1g + i * D_MODEL, ln1b + i * D_MODEL);
        // FFN1 + GELU
        k_gemm<<<dim3(MTOK / 128, D_FF / 64), 256, 0, stream>>>(
            Xh, Wt_c1 + (size_t)i * D_FF * D_MODEL, c1b + i * D_FF,
            nullptr, nullptr, Yh, D_FF, D_MODEL, 1);
        // FFN2 + residual
        k_gemm<<<dim3(MTOK / 128, D_MODEL / 64), 256, 0, stream>>>(
            Yh, Wt_c2 + (size_t)i * D_MODEL * D_FF, c2b + i * D_MODEL,
            X, X, nullptr, D_MODEL, D_FF, 0);
        k_ln<<<MTOK, 256, 0, stream>>>(X, Xh, ln2g + i * D_MODEL, ln2b + i * D_MODEL);
    }

    // ---------------- batch norm + heads ----------------
    k_zero<<<2, 256, 0, stream>>>(bns, D_MODEL);
    k_zero<<<2, 256, 0, stream>>>(bnq, D_MODEL);
    k_bnstats<<<MTOK / 128, 256, 0, stream>>>(X, bns, bnq);
    k_bnfin<<<2, 256, 0, stream>>>(bns, bnq, bng, bnb, bnsc, bnsh);
    k_head<<<BFLAT, 256, 0, stream>>>(X, bnsc, bnsh, h1w, h1b, means, stdev, valb);
    k_final<<<BS, 64, 0, stream>>>(valb, h2w, h2b, out);
}
